// AttentionHead_82738249990896
// MI455X (gfx1250) — matrix operand order repaired and checked by reading, not compiled, not run on hardware
//
#include <hip/hip_runtime.h>
#include <hip/hip_bf16.h>

#define B_ 16
#define T_ 2048
#define E_ 1024
#define H_ 128

typedef __attribute__((ext_vector_type(16))) __bf16 v16bf;
typedef __attribute__((ext_vector_type(8)))  float  v8f;

union BfPack {
  uint4 u4[2];
  unsigned int u[8];
  unsigned short s[16];
  v16bf v;
};

static __device__ __forceinline__ unsigned short f2bf(float f) {
  union { float f; unsigned int u; } c; c.f = f;
  unsigned int r = c.u + 0x7FFFu + ((c.u >> 16) & 1u);  // round-to-nearest-even
  return (unsigned short)(r >> 16);
}

// Two f32 -> packed bf16x2 in one v_perm_b32 (+1 add each for round-half-up).
static __device__ __forceinline__ unsigned int pack_bf16x2(float a, float b) {
  unsigned int ua = __float_as_uint(a) + 0x8000u;
  unsigned int ub = __float_as_uint(b) + 0x8000u;
  // {S0=ub bytes 4..7, S1=ua bytes 0..3}; sel 0x07060302 -> [ub3 ub2 ua3 ua2]
  return __builtin_amdgcn_perm(ub, ua, 0x07060302u);
}

static __device__ __forceinline__ v8f vzero8() {
  v8f z = {0.f, 0.f, 0.f, 0.f, 0.f, 0.f, 0.f, 0.f};
  return z;
}

// CDNA5 async copy: global -> LDS, 16B per lane, tracked by ASYNCcnt.
static __device__ __forceinline__ void async_ld_b128(void* lds_dst, const void* gsrc) {
  unsigned int       l = (unsigned int)(unsigned long long)(uintptr_t)lds_dst;  // addr[31:0] = LDS byte addr
  unsigned long long g = (unsigned long long)(uintptr_t)gsrc;
  asm volatile("global_load_async_to_lds_b128 %0, %1, off" :: "v"(l), "v"(g) : "memory");
}

static __device__ __forceinline__ void wait_asynccnt0() {
  asm volatile("s_wait_asynccnt 0" ::: "memory");
}

// ---------------------------------------------------------------------------
// Kernel 1: convert + transpose [Wq|Wk|Wv] (E x H each, f32) into bf16
// Wt[n][k], n in [0,384): 0-127 -> Wq, 128-255 -> Wk, 256-383 -> Wv.
// ---------------------------------------------------------------------------
__global__ void __launch_bounds__(256) prep_wt(const float* __restrict__ Wk,
                                               const float* __restrict__ Wq,
                                               const float* __restrict__ Wv,
                                               unsigned short* __restrict__ Wt) {
  int idx = blockIdx.x * 256 + threadIdx.x;  // 384*1024 total
  int n = idx >> 10;
  int k = idx & 1023;
  const float* W = (n < 128) ? Wq : ((n < 256) ? Wk : Wv);
  int col = n & 127;
  Wt[idx] = f2bf(W[k * H_ + col]);
}

// ---------------------------------------------------------------------------
// Kernel 2: Q,K,V projections. One wave -> 16x64 output tile, 32 WMMA steps
// over E=1024. A = x tile converted f32->bf16 via v_perm packing.
// Q,K stored [t][h]; V stored transposed Vt[b][h][t].
// ---------------------------------------------------------------------------
__global__ void __launch_bounds__(128) proj_qkv(const float* __restrict__ x,
                                                const unsigned short* __restrict__ Wt,
                                                unsigned short* __restrict__ Q,
                                                unsigned short* __restrict__ K,
                                                unsigned short* __restrict__ Vt) {
  const int wave = threadIdx.x >> 5;
  const int lane = threadIdx.x & 31;
  const int grp  = lane >> 4;
  const int nn   = lane & 15;

  const int mtile  = blockIdx.x * 4 + wave;   // 2048 M-tiles of 16 rows
  const int n0base = blockIdx.y * 64;         // 6 N-groups of 64 cols
  const int m0  = mtile * 16;
  const int row = m0 + nn;                    // A-operand: row = lane%16

  v8f acc0 = vzero8(), acc1 = vzero8(), acc2 = vzero8(), acc3 = vzero8();

  const float* xr = x + (size_t)row * E_ + grp * 16;
  const unsigned short* wr0 = Wt + (size_t)(n0base +  0 + nn) * E_ + grp * 16;
  const unsigned short* wr1 = Wt + (size_t)(n0base + 16 + nn) * E_ + grp * 16;
  const unsigned short* wr2 = Wt + (size_t)(n0base + 32 + nn) * E_ + grp * 16;
  const unsigned short* wr3 = Wt + (size_t)(n0base + 48 + nn) * E_ + grp * 16;

  #pragma unroll 4
  for (int it = 0; it < 32; ++it) {
    const float* xp = xr + it * 32;
    float4 a0 = *reinterpret_cast<const float4*>(xp);
    float4 a1 = *reinterpret_cast<const float4*>(xp + 4);
    float4 a2 = *reinterpret_cast<const float4*>(xp + 8);
    float4 a3 = *reinterpret_cast<const float4*>(xp + 12);
    BfPack A;
    A.u[0] = pack_bf16x2(a0.x, a0.y);
    A.u[1] = pack_bf16x2(a0.z, a0.w);
    A.u[2] = pack_bf16x2(a1.x, a1.y);
    A.u[3] = pack_bf16x2(a1.z, a1.w);
    A.u[4] = pack_bf16x2(a2.x, a2.y);
    A.u[5] = pack_bf16x2(a2.z, a2.w);
    A.u[6] = pack_bf16x2(a3.x, a3.y);
    A.u[7] = pack_bf16x2(a3.z, a3.w);

    BfPack B0, B1, B2, B3;
    B0.u4[0] = *reinterpret_cast<const uint4*>(wr0 + it * 32);
    B0.u4[1] = *reinterpret_cast<const uint4*>(wr0 + it * 32 + 8);
    B1.u4[0] = *reinterpret_cast<const uint4*>(wr1 + it * 32);
    B1.u4[1] = *reinterpret_cast<const uint4*>(wr1 + it * 32 + 8);
    B2.u4[0] = *reinterpret_cast<const uint4*>(wr2 + it * 32);
    B2.u4[1] = *reinterpret_cast<const uint4*>(wr2 + it * 32 + 8);
    B3.u4[0] = *reinterpret_cast<const uint4*>(wr3 + it * 32);
    B3.u4[1] = *reinterpret_cast<const uint4*>(wr3 + it * 32 + 8);

    acc0 = __builtin_amdgcn_wmma_f32_16x16x32_bf16(false, A.v, false, B0.v, (short)0, acc0, false, false);
    acc1 = __builtin_amdgcn_wmma_f32_16x16x32_bf16(false, A.v, false, B1.v, (short)0, acc1, false, false);
    acc2 = __builtin_amdgcn_wmma_f32_16x16x32_bf16(false, A.v, false, B2.v, (short)0, acc2, false, false);
    acc3 = __builtin_amdgcn_wmma_f32_16x16x32_bf16(false, A.v, false, B3.v, (short)0, acc3, false, false);
  }

  v8f accs[4] = {acc0, acc1, acc2, acc3};
  #pragma unroll
  for (int i = 0; i < 4; ++i) {
    const int n0  = n0base + i * 16;
    const int col = n0 + nn;                   // C/D layout: N = lane%16
    if (col < 256) {                           // Q or K region: [t][h]
      unsigned short* dst = (col < 128) ? Q : K;
      const int h = col & 127;
      #pragma unroll
      for (int r = 0; r < 8; ++r) {            // M = r + 8*grp
        const int m = m0 + r + 8 * grp;
        dst[(size_t)m * H_ + h] = f2bf(accs[i][r]);
      }
    } else {                                   // V region: transposed [b][h][t]
      const int h  = col - 256;
      const int b  = m0 / T_;
      const int tl = (m0 & (T_ - 1)) + 8 * grp;
      unsigned int tmp[4];
      #pragma unroll
      for (int r = 0; r < 4; ++r) tmp[r] = pack_bf16x2(accs[i][2 * r], accs[i][2 * r + 1]);
      *reinterpret_cast<uint4*>(Vt + (size_t)b * H_ * T_ + (size_t)h * T_ + tl) =
          *reinterpret_cast<const uint4*>(tmp);
    }
  }
}

// ---------------------------------------------------------------------------
// Kernel 3: fused causal flash attention, one wave per workgroup.
// Per 32-key block: 8 WMMA (S=Q*K^T), online softmax, P via LDS transpose,
// 8 WMMA (out += P*V). K and V blocks are double-buffered in LDS via
// GLOBAL_LOAD_ASYNC_TO_LDS_B128 (ASYNCcnt), overlapping the next block's
// global traffic with the current block's WMMAs.
// ---------------------------------------------------------------------------
__global__ void __launch_bounds__(32) attn_fused(const unsigned short* __restrict__ Q,
                                                 const unsigned short* __restrict__ K,
                                                 const unsigned short* __restrict__ Vt,
                                                 float* __restrict__ out) {
  __shared__ unsigned short ldsK[2][32 * 128];  // 2 x 8 KB: 32 keys x 128 h
  __shared__ unsigned short ldsV[2][128 * 32];  // 2 x 8 KB: 128 h x 32 keys
  __shared__ unsigned short ldsP[16 * 32];      // 1 KB P staging

  const int lane = threadIdx.x & 31;
  const int grp  = lane >> 4;
  const int nn   = lane & 15;

  const int b  = blockIdx.y;
  const int t0 = blockIdx.x * 16;

  const unsigned short* Qb  = Q  + (size_t)b * T_ * H_;
  const unsigned short* Kb  = K  + (size_t)b * T_ * H_;
  const unsigned short* Vtb = Vt + (size_t)b * H_ * T_;

  // Resident Q tile as 4 A-operands (H=128 = 4 K32 chunks).
  v16bf Aq[4];
  const int qrow = t0 + nn;
  #pragma unroll
  for (int j = 0; j < 4; ++j) {
    const unsigned short* qp = Qb + (size_t)qrow * H_ + 32 * j + grp * 16;
    BfPack u;
    u.u4[0] = *reinterpret_cast<const uint4*>(qp);
    u.u4[1] = *reinterpret_cast<const uint4*>(qp + 8);
    Aq[j] = u.v;
  }

  v8f acc[8];
  #pragma unroll
  for (int c = 0; c < 8; ++c) acc[c] = vzero8();
  float mrow[8], ssum[8];
  #pragma unroll
  for (int r = 0; r < 8; ++r) { mrow[r] = -INFINITY; ssum[r] = 0.f; }

  // base-2 domain: scale2 = (1/sqrt(H)) * log2(e); exp2f -> native v_exp_f32
  const float scale2 = 0.0883883476483184f * 1.44269504088896340736f;
  const int nblocks = (t0 + 47) >> 5;  // keys [0, t0+16) in blocks of 32

  // Stage block 0 (K: 32x128 bf16, V: 128x32 bf16), 16B per lane per issue.
  {
    #pragma unroll
    for (int i = 0; i < 16; ++i) {
      const int off = (i * 32 + lane) * 8;
      async_ld_b128(&ldsK[0][off], Kb + off);
    }
    #pragma unroll
    for (int i = 0; i < 16; ++i) {
      const int c = i * 32 + lane;
      const int h = c >> 2, q = c & 3;
      async_ld_b128(&ldsV[0][h * 32 + q * 8], Vtb + (size_t)h * T_ + q * 8);
    }
  }

  for (int sb = 0; sb < nblocks; ++sb) {
    const int s0 = sb * 32;

    wait_asynccnt0();  // block sb resident in buf[sb&1]

    if (sb + 1 < nblocks) {  // stream block sb+1 while computing sb
      const int sn = s0 + 32;
      unsigned short* bK = ldsK[(sb + 1) & 1];
      unsigned short* bV = ldsV[(sb + 1) & 1];
      #pragma unroll
      for (int i = 0; i < 16; ++i) {
        const int off = (i * 32 + lane) * 8;
        async_ld_b128(bK + off, Kb + (size_t)sn * H_ + off);
      }
      #pragma unroll
      for (int i = 0; i < 16; ++i) {
        const int c = i * 32 + lane;
        const int h = c >> 2, q = c & 3;
        async_ld_b128(bV + h * 32 + q * 8, Vtb + (size_t)h * T_ + sn + q * 8);
      }
    }

    const unsigned short* bufK = ldsK[sb & 1];
    const unsigned short* bufV = ldsV[sb & 1];

    // S = Q * K^T for 32 keys -> two 16x16 f32 tiles (B operands from LDS)
    v8f S0 = vzero8(), S1 = vzero8();
    #pragma unroll
    for (int j = 0; j < 4; ++j) {
      const unsigned short* kp0 = bufK + (nn)      * H_ + 32 * j + grp * 16;
      const unsigned short* kp1 = bufK + (16 + nn) * H_ + 32 * j + grp * 16;
      BfPack B0, B1;
      B0.u4[0] = *reinterpret_cast<const uint4*>(kp0);
      B0.u4[1] = *reinterpret_cast<const uint4*>(kp0 + 8);
      B1.u4[0] = *reinterpret_cast<const uint4*>(kp1);
      B1.u4[1] = *reinterpret_cast<const uint4*>(kp1 + 8);
      S0 = __builtin_amdgcn_wmma_f32_16x16x32_bf16(false, Aq[j], false, B0.v, (short)0, S0, false, false);
      S1 = __builtin_amdgcn_wmma_f32_16x16x32_bf16(false, Aq[j], false, B1.v, (short)0, S1, false, false);
    }

    // Causal mask only on diagonal-straddling blocks.
    const bool edge = (s0 + 31 > t0);

    #pragma unroll
    for (int r = 0; r < 8; ++r) {
      float x0 = S0[r] * scale2;
      float x1 = S1[r] * scale2;
      if (edge) {
        const int trow = t0 + r + 8 * grp;
        if (s0 + nn      > trow) x0 = -INFINITY;
        if (s0 + 16 + nn > trow) x1 = -INFINITY;
      }

      float mx = fmaxf(x0, x1);
      mx = fmaxf(mx, __shfl_xor(mx, 1, 16));
      mx = fmaxf(mx, __shfl_xor(mx, 2, 16));
      mx = fmaxf(mx, __shfl_xor(mx, 4, 16));
      mx = fmaxf(mx, __shfl_xor(mx, 8, 16));

      const float mnew  = fmaxf(mrow[r], mx);
      const float alpha = exp2f(mrow[r] - mnew);
      mrow[r] = mnew;

      const float p0 = exp2f(x0 - mnew);
      const float p1 = exp2f(x1 - mnew);
      float rs = p0 + p1;
      rs += __shfl_xor(rs, 1, 16);
      rs += __shfl_xor(rs, 2, 16);
      rs += __shfl_xor(rs, 4, 16);
      rs += __shfl_xor(rs, 8, 16);
      ssum[r] = ssum[r] * alpha + rs;

      #pragma unroll
      for (int c = 0; c < 8; ++c) acc[c][r] *= alpha;

      // P (C-layout) -> LDS [row][col]
      const int prow = r + 8 * grp;
      ldsP[prow * 32 + nn]      = f2bf(p0);
      ldsP[prow * 32 + 16 + nn] = f2bf(p1);
    }

    // Same-wave LDS RAW on P staging.
    asm volatile("s_wait_dscnt 0" ::: "memory");

    // Reload P in A-operand layout: row = lane%16, 16 k-values per lane.
    BfPack up;
    const unsigned short* lp = ldsP + nn * 32 + grp * 16;
    up.u4[0] = *reinterpret_cast<const uint4*>(lp);
    up.u4[1] = *reinterpret_cast<const uint4*>(lp + 8);
    const v16bf Pa = up.v;

    // out += P * V (8 h-chunks of 16); V^T rows give contiguous B loads.
    #pragma unroll
    for (int c = 0; c < 8; ++c) {
      const unsigned short* vp = bufV + (16 * c + nn) * 32 + grp * 16;
      BfPack Bv;
      Bv.u4[0] = *reinterpret_cast<const uint4*>(vp);
      Bv.u4[1] = *reinterpret_cast<const uint4*>(vp + 8);
      acc[c] = __builtin_amdgcn_wmma_f32_16x16x32_bf16(false, Pa, false, Bv.v, (short)0, acc[c], false, false);
    }
  }

  // Epilogue: normalize and store f32 output [b][t][h].
  #pragma unroll
  for (int r = 0; r < 8; ++r) {
    const float inv = 1.0f / ssum[r];
    const int t = t0 + r + 8 * grp;
    float* op = out + ((size_t)b * T_ + t) * H_ + nn;
    #pragma unroll
    for (int c = 0; c < 8; ++c) op[16 * c] = acc[c][r] * inv;
  }
}

// ---------------------------------------------------------------------------
// Launch. Inputs (setup_inputs order): x, mask, Wk, Wq, Wv. mask is implicit
// (causal). Workspace: Wt bf16 (768 KB), Q/K bf16 (8 MB each), Vt bf16 (8 MB).
// ---------------------------------------------------------------------------
extern "C" void kernel_launch(void* const* d_in, const int* in_sizes, int n_in,
                              void* d_out, int out_size, void* d_ws, size_t ws_size,
                              hipStream_t stream) {
  (void)in_sizes; (void)n_in; (void)out_size; (void)ws_size;
  const float* x  = (const float*)d_in[0];
  const float* Wk = (const float*)d_in[2];
  const float* Wq = (const float*)d_in[3];
  const float* Wv = (const float*)d_in[4];
  float* out = (float*)d_out;

  char* ws = (char*)d_ws;
  const size_t wtBytes  = (size_t)384 * E_ * 2;          // 768 KB
  const size_t qkvBytes = (size_t)B_ * T_ * H_ * 2;      // 8 MB each
  unsigned short* Wt = (unsigned short*)(ws);
  unsigned short* Q  = (unsigned short*)(ws + wtBytes);
  unsigned short* K  = (unsigned short*)(ws + wtBytes + qkvBytes);
  unsigned short* Vt = (unsigned short*)(ws + wtBytes + 2 * qkvBytes);

  prep_wt<<<1536, 256, 0, stream>>>(Wk, Wq, Wv, Wt);
  proj_qkv<<<dim3((B_ * T_ / 16) / 4, 6), 128, 0, stream>>>(x, Wt, Q, K, Vt);
  attn_fused<<<dim3(T_ / 16, B_), 32, 0, stream>>>(Q, K, Vt, out);
}
